// WeightedCausalMultiheadAttention_63505386438809
// MI455X (gfx1250) — matrix operand order repaired and checked by reading, not compiled, not run on hardware
//
#include <hip/hip_runtime.h>
#include <hip/hip_bf16.h>

// ---------------------------------------------------------------------------
// WeightedCausalMultiheadAttention for MI455X (gfx1250), wave32 + WMMA bf16.
//   B=2, S=2048, D=1024, H=16, hd=64, ALPHA=1.
// Pipeline:
//   cvt (hidden f32->bf16), transpose-cvt (w_attn, w_proj -> [N][K] bf16)
//   -> QKV GEMM (WMMA 8/wave/step, async-to-LDS double buffering; V^T epilogue)
//   -> flash attention (WMMA; K/V blocks cooperatively staged via async-to-LDS)
//   -> output proj GEMM (WMMA, fp32 epilogue).
// Workspace (bytes), total 48 MiB:
//   [0,        8388608)  hidden bf16        [4096 x 1024]
//   [8388608, 14680064)  w_attn^T bf16      [3072 x 1024]
//   [14680064,16777216)  w_proj^T bf16      [1024 x 1024]
//   [16777216,25165824)  Q bf16             [B,H,S,hd]
//   [25165824,33554432)  K bf16             [B,H,S,hd]
//   [33554432,41943040)  V^T bf16           [B,H,hd,S]
//   [41943040,50331648)  attn_out bf16      [4096 x 1024]
// ---------------------------------------------------------------------------

#define S_LEN   2048
#define BATCH   2
#define DMODEL  1024
#define NHEADS  16
#define HDIM    64
#define ROWS    (BATCH * S_LEN)   // 4096
#define N3D     (3 * DMODEL)      // 3072
#define LDP     40                // padded stride (halves): 16B-aligned frags, 20-dword
                                  // stride -> conflict-free across 16 lanes
#define LDK     72                // K-tile stride (halves): 36-dword, conflict-free

typedef __attribute__((ext_vector_type(16))) __bf16 v16bf;
typedef __attribute__((ext_vector_type(8)))  float  v8f;

union Frag {
    v16bf          v;
    uint4          q[2];
    unsigned int   u[8];
    unsigned short s[16];
};

__device__ __forceinline__ unsigned short f2bf(float f) {
    unsigned int b = __float_as_uint(f);
    b += 0x7FFFu + ((b >> 16) & 1u);   // round-to-nearest-even
    return (unsigned short)(b >> 16);
}

__device__ __forceinline__ float half_reduce_max(float v) {
    #pragma unroll
    for (int m = 1; m < 16; m <<= 1) v = fmaxf(v, __shfl_xor(v, m, 32));
    return v;   // independent within lanes 0-15 / 16-31
}

__device__ __forceinline__ float half_reduce_sum(float v) {
    #pragma unroll
    for (int m = 1; m < 16; m <<= 1) v += __shfl_xor(v, m, 32);
    return v;
}

// mask: delta<0 -> -inf ; delta>0 -> -log(delta+1) ; delta==0 -> 0   (ALPHA=1)
__device__ __forceinline__ float apply_mask(float s, int delta) {
    if (delta < 0) return -1.0e30f;
    if (delta == 0) return s;
    return s - __logf((float)delta + 1.0f);
}

// A-matrix fragment: lane half hs needs K in [8hs, 8hs+7] and [16+8hs, 23+8hs].
__device__ __forceinline__ void load_afrag(Frag& f, const unsigned short* row, int hs) {
    f.q[0] = *(const uint4*)(row + 8 * hs);
    f.q[1] = *(const uint4*)(row + 16 + 8 * hs);
}

// B-matrix fragment from an N-major row (contiguous along K): lane half hs needs K in [8hs, 8hs+7] and [16+8hs, 23+8hs].
__device__ __forceinline__ void load_bfrag(Frag& f, const unsigned short* nrow, int hs) {
    f.q[0] = *(const uint4*)(nrow + 8 * hs);
    f.q[1] = *(const uint4*)(nrow + 16 + 8 * hs);
}

// CDNA5 async DMA: global -> LDS, 16B per lane, tracked by ASYNCcnt.
__device__ __forceinline__ void async_load_b128(const unsigned short* gptr, unsigned short* lptr) {
    unsigned lds_off = (unsigned)(unsigned long long)(uintptr_t)lptr;  // low 32 bits = LDS offset
    asm volatile("global_load_async_to_lds_b128 %0, %1, off"
                 :: "v"(lds_off), "v"((unsigned long long)(uintptr_t)gptr)
                 : "memory");
}
__device__ __forceinline__ void wait_async0() {
    asm volatile("s_wait_asynccnt 0x0" ::: "memory");
}

// ---------------------------------------------------------------------------
__global__ __launch_bounds__(256) void cvt_f32_bf16(const float* __restrict__ src,
                                                    unsigned short* __restrict__ dst,
                                                    int n) {
    int i      = blockIdx.x * blockDim.x + threadIdx.x;
    int stride = gridDim.x * blockDim.x;
    for (; i < n; i += stride) dst[i] = f2bf(src[i]);
}

// Transpose-convert: src f32 [R][C] -> dst bf16 [C][R].  R,C multiples of 32.
__global__ __launch_bounds__(256) void cvt_transpose_bf16(const float* __restrict__ src,
                                                          unsigned short* __restrict__ dst,
                                                          int R, int C) {
    __shared__ unsigned short tile[32][33];
    const int tx = threadIdx.x & 31, ty = threadIdx.x >> 5;   // ty in [0,8)
    const int r0 = blockIdx.y * 32, c0 = blockIdx.x * 32;
    #pragma unroll
    for (int i = 0; i < 32; i += 8)
        tile[ty + i][tx] = f2bf(src[(size_t)(r0 + ty + i) * C + c0 + tx]);
    __syncthreads();
    #pragma unroll
    for (int i = 0; i < 32; i += 8)
        dst[(size_t)(c0 + ty + i) * R + r0 + tx] = tile[tx][ty + i];
}

// ---------------------------------------------------------------------------
// GEMM core: C[ROWS, N] = A[ROWS,1024] x B^T[N,1024]  (B given N-major).
// Block tile 128(M) x 128(N); 8 waves in 2(M) x 4(N); 64x32 (4x2 WMMA) per wave.
// Async global->LDS double buffering, one barrier per 32-deep K step,
// 8 WMMA per wave per step fed by 12 ds_load_b128.
// ---------------------------------------------------------------------------

#define GEMM_PROLOG()                                                           \
    __shared__ unsigned short lA[2][128 * LDP];                                 \
    __shared__ unsigned short lB[2][128 * LDP];                                 \
    const int tid  = threadIdx.x;                                               \
    const int lane = tid & 31, w = tid >> 5;                                    \
    const int wm = w >> 2, wn = w & 3;                                          \
    const int ln = lane & 15, hs = lane >> 4;                                   \
    const int M0 = blockIdx.y * 128;                                            \
    const int N0 = blockIdx.x * 128;                                            \
    const int sr = tid >> 1, sk = (tid & 1) * 16;  /* 16 halves/thread/tile */  \
    const unsigned short* gA = A  + (size_t)(M0 + sr) * DMODEL + sk;            \
    const unsigned short* gB = Bt + (size_t)(N0 + sr) * DMODEL + sk;            \
    v8f acc[4][2];                                                              \
    _Pragma("unroll") for (int i = 0; i < 4; ++i)                               \
        _Pragma("unroll") for (int j = 0; j < 2; ++j)                           \
            acc[i][j] = (v8f){0.f,0.f,0.f,0.f,0.f,0.f,0.f,0.f};                 \
    /* stage tile 0 via async DMA */                                            \
    async_load_b128(gA,     &lA[0][sr * LDP + sk]);                             \
    async_load_b128(gA + 8, &lA[0][sr * LDP + sk + 8]);                         \
    async_load_b128(gB,     &lB[0][sr * LDP + sk]);                             \
    async_load_b128(gB + 8, &lB[0][sr * LDP + sk + 8]);                         \
    wait_async0();                                                              \
    __syncthreads();                                                            \
    int buf = 0;                                                                \
    for (int k0 = 0; k0 < DMODEL; k0 += 32) {                                   \
        const bool last = (k0 + 32 >= DMODEL);                                  \
        if (!last) {                                                            \
            async_load_b128(gA + k0 + 32, &lA[buf ^ 1][sr * LDP + sk]);         \
            async_load_b128(gA + k0 + 40, &lA[buf ^ 1][sr * LDP + sk + 8]);     \
            async_load_b128(gB + k0 + 32, &lB[buf ^ 1][sr * LDP + sk]);         \
            async_load_b128(gB + k0 + 40, &lB[buf ^ 1][sr * LDP + sk + 8]);     \
        }                                                                       \
        Frag fa[4], fb[2];                                                      \
        _Pragma("unroll")                                                       \
        for (int mt = 0; mt < 4; ++mt)                                          \
            load_afrag(fa[mt], &lA[buf][(wm * 64 + mt * 16 + ln) * LDP], hs);   \
        _Pragma("unroll")                                                       \
        for (int nt = 0; nt < 2; ++nt)                                          \
            load_bfrag(fb[nt], &lB[buf][(wn * 32 + nt * 16 + ln) * LDP], hs);   \
        _Pragma("unroll")                                                       \
        for (int mt = 0; mt < 4; ++mt)                                          \
            _Pragma("unroll")                                                   \
            for (int nt = 0; nt < 2; ++nt)                                      \
                acc[mt][nt] = __builtin_amdgcn_wmma_f32_16x16x32_bf16(          \
                    false, fa[mt].v, false, fb[nt].v, (short)0, acc[mt][nt],    \
                    false, false);                                              \
        if (!last) wait_async0();                                               \
        __syncthreads();                                                        \
        buf ^= 1;                                                               \
    }

// QKV GEMM: N = 3072; epilogue scatters into Q,K ([B,H,S,hd]) and V^T ([B,H,hd,S]).
__global__ __launch_bounds__(256) void gemm_qkv(const unsigned short* __restrict__ A,
                                                const unsigned short* __restrict__ Bt,
                                                const float* __restrict__ bias,
                                                unsigned short* __restrict__ Qo,
                                                unsigned short* __restrict__ Ko,
                                                unsigned short* __restrict__ Vt) {
    GEMM_PROLOG()

    #pragma unroll
    for (int nt = 0; nt < 2; ++nt) {
        const int cg   = N0 + wn * 32 + nt * 16 + ln;   // [0, 3072)
        const int part = cg / DMODEL;                   // 0=q 1=k 2=v
        const int dcol = cg % DMODEL;
        const int head = dcol >> 6, hd = dcol & 63;
        const float bv = bias[cg];
        #pragma unroll
        for (int mt = 0; mt < 4; ++mt) {
            #pragma unroll
            for (int j = 0; j < 8; ++j) {
                const int rg = M0 + wm * 64 + mt * 16 + j + (hs << 3);
                const int bi = rg >> 11, si = rg & (S_LEN - 1);
                const int bh = bi * NHEADS + head;
                const unsigned short val = f2bf(acc[mt][nt][j] + bv);
                if (part == 0)
                    Qo[((size_t)bh * S_LEN + si) * HDIM + hd] = val;
                else if (part == 1)
                    Ko[((size_t)bh * S_LEN + si) * HDIM + hd] = val;
                else
                    Vt[((size_t)bh * HDIM + hd) * S_LEN + si] = val;   // transposed
            }
        }
    }
}

// Output projection: N = 1024; fp32 epilogue + bias into d_out.
__global__ __launch_bounds__(256) void gemm_proj(const unsigned short* __restrict__ A,
                                                 const unsigned short* __restrict__ Bt,
                                                 const float* __restrict__ bias,
                                                 float* __restrict__ out) {
    GEMM_PROLOG()

    #pragma unroll
    for (int nt = 0; nt < 2; ++nt) {
        const int cg = N0 + wn * 32 + nt * 16 + ln;
        const float bv = bias[cg];
        #pragma unroll
        for (int mt = 0; mt < 4; ++mt) {
            #pragma unroll
            for (int j = 0; j < 8; ++j) {
                const int rg = M0 + wm * 64 + mt * 16 + j + (hs << 3);
                out[(size_t)rg * DMODEL + cg] = acc[mt][nt][j] + bv;
            }
        }
    }
}

// ---------------------------------------------------------------------------
// Flash attention (forward-looking power-law mask).
// Workgroup = 128 queries (8 waves x 16), one (b,h). All waves walk the same
// 32-key blocks; K (32x64) and V^T (64x32) are staged cooperatively into LDS
// via async DMA with double buffering (one barrier per block). Waves whose
// queries lie entirely after the key block skip the (fully masked) compute.
__global__ __launch_bounds__(256) void attn_fwd(const unsigned short* __restrict__ Q,
                                                const unsigned short* __restrict__ K,
                                                const unsigned short* __restrict__ Vt,
                                                unsigned short* __restrict__ O) {
    __shared__ unsigned short lK[2][32 * LDK];    // [key][hd]   9216 B
    __shared__ unsigned short lV[2][64 * LDP];    // [hd][key]  10240 B
    __shared__ unsigned short lP[8 * 16 * LDP];   // per-wave P 10240 B

    const int tid  = threadIdx.x;
    const int lane = tid & 31, w = tid >> 5;
    const int ln = lane & 15, hs = lane >> 4;
    const int bh = blockIdx.y;                    // b*H + h
    const int h = bh & (NHEADS - 1), b = bh >> 4;
    const int Q0 = blockIdx.x * 128;              // 32-aligned
    const int q0 = Q0 + w * 16;
    const size_t base = (size_t)bh * S_LEN * HDIM;
    unsigned short* wp = &lP[w * 16 * LDP];

    // cooperative staging maps (16B per thread per matrix per block)
    const int kr = tid >> 3, kc = (tid & 7) * 8;  // K: row=key [0,32), col=hd
    const int vr = tid >> 2, vc = (tid & 3) * 8;  // V: row=hd [0,64), col=key
    const unsigned short* gK = K  + base + (size_t)kr * HDIM + kc;   // + j0*HDIM
    const unsigned short* gV = Vt + base + (size_t)vr * S_LEN + vc;  // + j0

    // Q A-fragments, two 32-deep k-chunks covering hd=64
    Frag qf0, qf1;
    {
        const unsigned short* qrow = Q + base + (size_t)(q0 + ln) * HDIM;
        load_afrag(qf0, qrow, hs);
        load_afrag(qf1, qrow + 32, hs);
    }

    v8f oacc[4];
    #pragma unroll
    for (int c = 0; c < 4; ++c) oacc[c] = (v8f){0.f,0.f,0.f,0.f,0.f,0.f,0.f,0.f};
    float m8[8], l8[8];
    #pragma unroll
    for (int j = 0; j < 8; ++j) { m8[j] = -1.0e30f; l8[j] = 0.f; }

    const float scale = 0.125f;   // hd^-0.5

    // stage key block 0
    async_load_b128(gK + (size_t)Q0 * HDIM, &lK[0][kr * LDK + kc]);
    async_load_b128(gV + Q0,                &lV[0][vr * LDP + vc]);
    wait_async0();
    __syncthreads();

    int buf = 0;
    for (int j0 = Q0; j0 < S_LEN; j0 += 32) {
        const bool last = (j0 + 32 >= S_LEN);
        if (!last) {
            async_load_b128(gK + (size_t)(j0 + 32) * HDIM, &lK[buf ^ 1][kr * LDK + kc]);
            async_load_b128(gV + (j0 + 32),                &lV[buf ^ 1][vr * LDP + vc]);
        }

        if (j0 + 32 > q0) {   // wave-uniform: skip fully masked blocks
            // --- scores: 16 queries x 32 keys from staged K ---
            v8f s0 = {0.f,0.f,0.f,0.f,0.f,0.f,0.f,0.f};
            v8f s1 = {0.f,0.f,0.f,0.f,0.f,0.f,0.f,0.f};
            {
                const unsigned short* krow0 = &lK[buf][ln * LDK];
                const unsigned short* krow1 = &lK[buf][(16 + ln) * LDK];
                Frag k00, k01, k10, k11;
                load_bfrag(k00, krow0, hs);        // hd 0..31
                load_bfrag(k01, krow0 + 32, hs);   // hd 32..63
                load_bfrag(k10, krow1, hs);
                load_bfrag(k11, krow1 + 32, hs);
                s0 = __builtin_amdgcn_wmma_f32_16x16x32_bf16(false, qf0.v, false, k00.v, (short)0, s0, false, false);
                s0 = __builtin_amdgcn_wmma_f32_16x16x32_bf16(false, qf1.v, false, k01.v, (short)0, s0, false, false);
                s1 = __builtin_amdgcn_wmma_f32_16x16x32_bf16(false, qf0.v, false, k10.v, (short)0, s1, false, false);
                s1 = __builtin_amdgcn_wmma_f32_16x16x32_bf16(false, qf1.v, false, k11.v, (short)0, s1, false, false);
            }

            // --- online softmax (row = query, spread across a 16-lane half) ---
            #pragma unroll
            for (int j = 0; j < 8; ++j) {
                const int row = q0 + j + (hs << 3);
                float sv0 = apply_mask(s0[j] * scale, (j0 + ln) - row);
                float sv1 = apply_mask(s1[j] * scale, (j0 + 16 + ln) - row);
                float rm   = half_reduce_max(fmaxf(sv0, sv1));
                float mnew = fmaxf(m8[j], rm);
                float corr = __expf(m8[j] - mnew);
                float p0 = __expf(sv0 - mnew);
                float p1 = __expf(sv1 - mnew);
                l8[j] = l8[j] * corr + half_reduce_sum(p0 + p1);
                m8[j] = mnew;
                #pragma unroll
                for (int c = 0; c < 4; ++c) oacc[c][j] *= corr;
                // C-layout -> LDS (row-major 16 x 32, padded), wave-private
                wp[(j + (hs << 3)) * LDP + ln]      = f2bf(p0);
                wp[(j + (hs << 3)) * LDP + 16 + ln] = f2bf(p1);
            }
            asm volatile("" ::: "memory");   // keep LDS store->load ordered

            // --- reload P as A-fragment (two b128s) ---
            Frag pf;
            load_afrag(pf, &wp[ln * LDP], hs);

            // --- O += P * V from staged V^T (rows contiguous along keys) ---
            #pragma unroll
            for (int c = 0; c < 4; ++c) {
                Frag vf;
                load_bfrag(vf, &lV[buf][(c * 16 + ln) * LDP], hs);
                oacc[c] = __builtin_amdgcn_wmma_f32_16x16x32_bf16(false, pf.v, false, vf.v,
                                                                  (short)0, oacc[c], false, false);
            }
        }

        if (!last) wait_async0();
        __syncthreads();
        buf ^= 1;
    }

    // --- normalize + store merged-head layout [B,S,D] bf16 ---
    #pragma unroll
    for (int j = 0; j < 8; ++j) {
        const float inv = 1.0f / l8[j];
        const int q = q0 + j + (hs << 3);
        const size_t ob = ((size_t)b * S_LEN + q) * DMODEL + h * HDIM;
        O[ob + 0  + ln] = f2bf(oacc[0][j] * inv);
        O[ob + 16 + ln] = f2bf(oacc[1][j] * inv);
        O[ob + 32 + ln] = f2bf(oacc[2][j] * inv);
        O[ob + 48 + ln] = f2bf(oacc[3][j] * inv);
    }
}

// ---------------------------------------------------------------------------
extern "C" void kernel_launch(void* const* d_in, const int* in_sizes, int n_in,
                              void* d_out, int out_size, void* d_ws, size_t ws_size,
                              hipStream_t stream) {
    (void)in_sizes; (void)n_in; (void)out_size; (void)ws_size;

    const float* f_hid = (const float*)d_in[0];   // [2,2048,1024]
    const float* f_wa  = (const float*)d_in[1];   // [1024,3072]
    const float* f_ba  = (const float*)d_in[2];   // [3072]
    const float* f_wp  = (const float*)d_in[3];   // [1024,1024]
    const float* f_bp  = (const float*)d_in[4];   // [1024]
    float*       out   = (float*)d_out;

    char* w = (char*)d_ws;
    unsigned short* hidB = (unsigned short*)(w + 0);
    unsigned short* waT  = (unsigned short*)(w + 8388608);    // [3072][1024]
    unsigned short* wpT  = (unsigned short*)(w + 14680064);   // [1024][1024]
    unsigned short* Qb   = (unsigned short*)(w + 16777216);
    unsigned short* Kb   = (unsigned short*)(w + 25165824);
    unsigned short* VbT  = (unsigned short*)(w + 33554432);   // [B,H,hd,S]
    unsigned short* AOb  = (unsigned short*)(w + 41943040);

    cvt_f32_bf16<<<dim3(1024), 256, 0, stream>>>(f_hid, hidB, ROWS * DMODEL);
    cvt_transpose_bf16<<<dim3(N3D / 32, DMODEL / 32), 256, 0, stream>>>(f_wa, waT, DMODEL, N3D);
    cvt_transpose_bf16<<<dim3(DMODEL / 32, DMODEL / 32), 256, 0, stream>>>(f_wp, wpT, DMODEL, DMODEL);

    gemm_qkv<<<dim3(N3D / 128, ROWS / 128), 256, 0, stream>>>(hidB, waT, f_ba, Qb, Kb, VbT);

    attn_fwd<<<dim3(S_LEN / 128, BATCH * NHEADS), 256, 0, stream>>>(Qb, Kb, VbT, AOb);

    gemm_proj<<<dim3(DMODEL / 128, ROWS / 128), 256, 0, stream>>>(AOb, wpT, f_bp, out);
}
